// Encoder_28595892256966
// MI455X (gfx1250) — matrix operand order repaired and checked by reading, not compiled, not run on hardware
//
#include <hip/hip_runtime.h>

// ---------------------------------------------------------------------------
// CDNA5 (gfx1250) tree point-cloud encoder.
// GEMMs: v_wmma_f32_16x16x32_bf16; fragments are unconditional b128 global
// loads (weights pre-transposed to [N,K], all operands zero-padded so
// K % 32 == 0, M/N edges handled by index clamping + guarded stores).
// Whole working set lives in the 192MB L2 -> no LDS staging needed.
// Per K-step: batch-load A + all 8 B fragments, then 8 back-to-back WMMAs;
// sched_group_barrier pins the [loads][WMMAs] shape so unrolled iterations
// overlap memory with matrix math.
// ---------------------------------------------------------------------------

typedef __attribute__((ext_vector_type(16))) __bf16 v16bf;
typedef __attribute__((ext_vector_type(8)))  __bf16 v8bf;
typedef __attribute__((ext_vector_type(8)))  float  v8f;

#define WG 256     // 8 wave32 waves per block
#define TM 128     // block tile M: 16 rows per wave
#define TN 128     // block tile N: 8 WMMA tiles per wave
#define TK 32      // K step == WMMA K

__device__ __forceinline__ unsigned short f2bf(float f) {
  unsigned int u = __float_as_uint(f);
  u += 0x7FFFu + ((u >> 16) & 1u);        // round-to-nearest-even
  return (unsigned short)(u >> 16);
}

__device__ __forceinline__ v16bf cat16(v8bf lo, v8bf hi) {
  return __builtin_shufflevector(lo, hi, 0,1,2,3,4,5,6,7,8,9,10,11,12,13,14,15);
}

// ---------------------------------------------------------------------------
// bf16 WMMA GEMM: C[M,N] = A[M,K] * Wt[N,K]^T (+ bias), fp32 accumulate.
// PRECONDITION: K % 32 == 0 (operands zero-padded by their producers).
// ---------------------------------------------------------------------------
__global__ __launch_bounds__(WG) void gemm_bf16_wmma(
    const unsigned short* __restrict__ A,    // [M,K]  bf16 bits
    const unsigned short* __restrict__ Wt,   // [N,K]  bf16 bits (pre-transposed)
    const float* __restrict__ bias,          // [N] or null
    float* __restrict__ C,                   // [M,N]  fp32
    int M, int N, int K)
{
  __builtin_amdgcn_s_wait_tensorcnt(0);      // CDNA5 split-counter wait (no-op)

  const int lane = threadIdx.x & 31;
  const int wave = threadIdx.x >> 5;
  const int m0 = blockIdx.y * TM;
  const int n0 = blockIdx.x * TN;
  const int l15 = lane & 15;

  // A fragment (16-bit A 16x32 ISA layout): lanes 0-15 hold K {0..7,16..23},
  // lanes 16-31 hold K {8..15,24..31}; row = lane&15 within the wave strip.
  const int kbA = (lane < 16) ? 0 : 8;
  // B fragment: same K order as A: lanes 0-15 hold K {0..7,16..23},
  // lanes 16-31 hold K {8..15,24..31} of their column.
  const int kbB = (lane < 16) ? 0 : 8;

  // Clamp OOB rows/cols to the last valid one: loads stay legal, the junk
  // lands in accumulators whose stores are guarded in the epilogue.
  const int rowA = min(m0 + 16 * wave + l15, M - 1);
  const unsigned short* __restrict__ Arow = A + (size_t)rowA * K + kbA;

  const unsigned short* __restrict__ Bcol[8];
  #pragma unroll
  for (int t = 0; t < 8; ++t) {
    int col = min(n0 + 16 * t + l15, N - 1);
    Bcol[t] = Wt + (size_t)col * K + kbB;
  }

  v8f acc[8] = {};

  for (int kk = 0; kk < K; kk += TK) {
    __builtin_prefetch(Arow + kk + TK, 0, 1);     // global_prefetch_b8

    // Batch ALL fragment loads for this K-step into independent registers.
    v16bf afrag = cat16(*(const v8bf*)(Arow + kk),
                        *(const v8bf*)(Arow + kk + 16));
    v16bf bfrag[8];
    #pragma unroll
    for (int t = 0; t < 8; ++t)
      bfrag[t] = cat16(*(const v8bf*)(Bcol[t] + kk),
                       *(const v8bf*)(Bcol[t] + kk + 16));

    #pragma unroll
    for (int t = 0; t < 8; ++t)
      acc[t] = __builtin_amdgcn_wmma_f32_16x16x32_bf16(
          /*neg_a=*/false, afrag, /*neg_b=*/false, bfrag[t],
          /*c_mod=*/(short)0, acc[t], /*reuse_a=*/false, /*reuse_b=*/false);

#if defined(__has_builtin)
#if __has_builtin(__builtin_amdgcn_sched_group_barrier)
    // Pin schedule per K-step: all VMEM reads first, then the 8 WMMAs.
    // With the compiler's 2x unroll this overlaps step k+1 loads with
    // step k matrix math instead of load->wait->WMMA chains.
    __builtin_amdgcn_sched_group_barrier(0x0020, 18, 0);  // VMEM read group
    __builtin_amdgcn_sched_group_barrier(0x0008,  8, 0);  // MFMA/WMMA group
#endif
#endif
  }

  // C/D layout: lane&15 = column, vgpr v = row (v + (lane<16?0:8))
  #pragma unroll
  for (int t = 0; t < 8; ++t) {
    int gn = n0 + 16 * t + l15;
    if (gn >= N) continue;
    float bv = bias ? bias[gn] : 0.0f;
    #pragma unroll
    for (int v = 0; v < 8; ++v) {
      int gm = m0 + 16 * wave + v + ((lane < 16) ? 0 : 8);
      if (gm < M) C[(size_t)gm * N + gn] = acc[t][v] + bv;
    }
  }
}

// ---------------------------------------------------------------------------
// Elementwise / reduction helpers
// ---------------------------------------------------------------------------
__global__ void k_f32_to_bf16(const float* __restrict__ x,
                              unsigned short* __restrict__ y, long long n) {
  long long i = (long long)blockIdx.x * blockDim.x + threadIdx.x;
  long long s = (long long)gridDim.x * blockDim.x;
  for (; i < n; i += s) y[i] = f2bf(x[i]);
}

// fp32 [M,K] -> bf16 [M,Kp], zero-padded columns K..Kp-1
__global__ void k_pack_rows_bf16(const float* __restrict__ src,
                                 unsigned short* __restrict__ dst,
                                 int K, int Kp, long long total) {
  long long i = (long long)blockIdx.x * blockDim.x + threadIdx.x;
  long long s = (long long)gridDim.x * blockDim.x;
  for (; i < total; i += s) {
    int c = (int)(i % Kp);
    long long r = i / Kp;
    dst[i] = (c < K) ? f2bf(src[r * K + c]) : (unsigned short)0;
  }
}

// fp32 [K,N] -> bf16 transposed [N,Kp], zero-padded
__global__ void k_convert_wT(const float* __restrict__ w,
                             unsigned short* __restrict__ wt,
                             int K, int Kp, int N, long long total) {
  long long i = (long long)blockIdx.x * blockDim.x + threadIdx.x;
  long long s = (long long)gridDim.x * blockDim.x;
  for (; i < total; i += s) {
    int k = (int)(i % Kp);
    int n = (int)(i / Kp);
    wt[i] = (k < K) ? f2bf(w[(size_t)k * N + n]) : (unsigned short)0;
  }
}

// training-mode BN stats over rows: one block per channel
__global__ __launch_bounds__(WG) void k_bn_stats(
    const float* __restrict__ X, float* __restrict__ mean,
    float* __restrict__ invstd, int M, int C) {
  int c = blockIdx.x;
  __shared__ float s1[WG], s2[WG];
  float a = 0.f, b = 0.f;
  for (int m = threadIdx.x; m < M; m += WG) {
    float v = X[(size_t)m * C + c];
    a += v; b += v * v;
  }
  s1[threadIdx.x] = a; s2[threadIdx.x] = b;
  __syncthreads();
  for (int s = WG / 2; s > 0; s >>= 1) {
    if (threadIdx.x < s) {
      s1[threadIdx.x] += s1[threadIdx.x + s];
      s2[threadIdx.x] += s2[threadIdx.x + s];
    }
    __syncthreads();
  }
  if (threadIdx.x == 0) {
    float mu  = s1[0] / (float)M;
    float var = s2[0] / (float)M - mu * mu;
    mean[c]   = mu;
    invstd[c] = rsqrtf(var + 1e-5f);
  }
}

// BN + activation, emits fp32 and/or bf16 (either pointer may be null).
// mode: 0 = none, 1 = ReLU, 2 = PReLU(alpha)
__global__ void k_bn_apply(const float* __restrict__ X,
                           float* __restrict__ Yf, unsigned short* __restrict__ Yb,
                           const float* __restrict__ mean, const float* __restrict__ invstd,
                           const float* __restrict__ g, const float* __restrict__ be,
                           long long n, int C, int mode, const float* __restrict__ alpha) {
  long long i = (long long)blockIdx.x * blockDim.x + threadIdx.x;
  long long s = (long long)gridDim.x * blockDim.x;
  float al = (mode == 2 && alpha) ? alpha[0] : 0.f;
  for (; i < n; i += s) {
    int c = (int)(i % C);
    float y = (X[i] - mean[c]) * (g[c] * invstd[c]) + be[c];
    if (mode == 1)      y = fmaxf(y, 0.f);
    else if (mode == 2) y = (y >= 0.f) ? y : al * y;
    if (Yf) Yf[i] = y;
    if (Yb) Yb[i] = f2bf(y);
  }
}

// fused BN + max over points for c3 (gamma>0 -> per-channel affine is monotone)
__global__ __launch_bounds__(WG) void k_bn_colmax(
    const float* __restrict__ H, float* __restrict__ G,
    const float* __restrict__ mean, const float* __restrict__ invstd,
    const float* __restrict__ g, const float* __restrict__ be,
    int Nn, int C) {
  int b = blockIdx.x;
  for (int c = threadIdx.x; c < C; c += blockDim.x) {
    float mx = -3.402823466e38f;
    const float* p = H + (size_t)b * Nn * C + c;
    for (int n = 0; n < Nn; ++n) mx = fmaxf(mx, p[(size_t)n * C]);
    G[b * C + c] = (mx - mean[c]) * (g[c] * invstd[c]) + be[c];
  }
}

// x' = p @ (T + I); one thread per point; writes bf16 row padded to Kp
__global__ void k_apply_transform(const float* __restrict__ pts,
                                  const float* __restrict__ t9,
                                  unsigned short* __restrict__ outbf,
                                  int Nn, int Kp, long long totalPts) {
  long long i = (long long)blockIdx.x * blockDim.x + threadIdx.x;
  long long s = (long long)gridDim.x * blockDim.x;
  for (; i < totalPts; i += s) {
    int b = (int)(i / Nn);
    const float* T = t9 + b * 9;
    float px = pts[i * 3], py = pts[i * 3 + 1], pz = pts[i * 3 + 2];
    float x = px * (T[0] + 1.f) + py * T[3] + pz * T[6];
    float y = px * T[1] + py * (T[4] + 1.f) + pz * T[7];
    float z = px * T[2] + py * T[5] + pz * (T[8] + 1.f);
    unsigned short* o = outbf + i * Kp;
    o[0] = f2bf(x); o[1] = f2bf(y); o[2] = f2bf(z);
    for (int c = 3; c < Kp; ++c) o[c] = 0;
  }
}

// per-batch permutation gather (leaf -> tree-leaf order)
__global__ void k_gather_arrange(const float* __restrict__ src, float* __restrict__ dst,
                                 const int* __restrict__ arr, int Nn, int C,
                                 long long total) {
  long long i = (long long)blockIdx.x * blockDim.x + threadIdx.x;
  long long s = (long long)gridDim.x * blockDim.x;
  for (; i < total; i += s) {
    int c = (int)(i % C);
    long long bj = i / C;
    int j = (int)(bj % Nn);
    int b = (int)(bj / Nn);
    dst[i] = src[((size_t)b * Nn + arr[b * Nn + j]) * C + c];
  }
}

// tree child gather + bf16 convert, dst rows padded to Kp
__global__ void k_gather_convert(const float* __restrict__ src,
                                 unsigned short* __restrict__ dst,
                                 const int* __restrict__ idx,
                                 int rows, int K, int Kp, int srcNodes,
                                 long long total) {
  long long i = (long long)blockIdx.x * blockDim.x + threadIdx.x;
  long long s = (long long)gridDim.x * blockDim.x;
  for (; i < total; i += s) {
    int c = (int)(i % Kp);
    long long bj = i / Kp;
    int j = (int)(bj % rows);
    int b = (int)(bj / rows);
    dst[i] = (c < K) ? f2bf(src[((size_t)b * srcNodes + idx[j]) * K + c])
                     : (unsigned short)0;
  }
}

// left/right child max + optional sampled-feature concat
__global__ void k_tree_combine(const float* __restrict__ h,    // [B, 2n, OUT]
                               const float* __restrict__ samp, // [B, snodes, sc] or null
                               const int* __restrict__ sidx,
                               float* __restrict__ out,        // [B, n, OUT+sc]
                               int n, int OUT, int sc, int snodes, long long total) {
  long long i = (long long)blockIdx.x * blockDim.x + threadIdx.x;
  long long s = (long long)gridDim.x * blockDim.x;
  int od = OUT + sc;
  for (; i < total; i += s) {
    int c = (int)(i % od);
    long long bi = i / od;
    int node = (int)(bi % n);
    int b = (int)(bi / n);
    if (c < OUT) {
      float a = h[((size_t)b * 2 * n + node)     * OUT + c];
      float d = h[((size_t)b * 2 * n + node + n) * OUT + c];
      out[i] = fmaxf(a, d);
    } else {
      out[i] = samp[((size_t)b * snodes + sidx[node]) * sc + (c - OUT)];
    }
  }
}

// ---------------------------------------------------------------------------
// Orchestration
// ---------------------------------------------------------------------------
extern "C" void kernel_launch(void* const* d_in, const int* in_sizes, int n_in,
                              void* d_out, int out_size, void* d_ws, size_t ws_size,
                              hipStream_t stream) {
  (void)in_sizes; (void)n_in; (void)out_size; (void)ws_size;
  const int B = 32, N = 1024;
  const long long NP = (long long)B * N;

  auto F = [&](int i) { return (const float*)d_in[i]; };
  auto I = [&](int i) { return (const int*)d_in[i]; };

  const float* points = F(0);
  const float *c1W=F(1),  *c1b=F(2),  *c1g=F(3),  *c1be=F(4);
  const float *c2W=F(5),  *c2b=F(6),  *c2g=F(7),  *c2be=F(8);
  const float *c3W=F(9),  *c3b=F(10), *c3g=F(11), *c3be=F(12);
  const float *f1W=F(13), *f1b=F(14), *f1g=F(15), *f1be=F(16);
  const float *f2W=F(17), *f2b=F(18), *f2g=F(19), *f2be=F(20);
  const float *f3W=F(21), *f3b=F(22);
  const float *l0W=F(23), *l0b=F(24), *l0g=F(25), *l0be=F(26), *l0a=F(27);
  const float *l1W=F(28), *l1b=F(29), *l1g=F(30), *l1be=F(31);
  const float *upW[10], *upb[10], *upg[10], *upbe[10];
  for (int k = 0; k < 10; ++k) {
    upW[k]=F(32+4*k); upb[k]=F(33+4*k); upg[k]=F(34+4*k); upbe[k]=F(35+4*k);
  }
  const int* arrange = I(72);
  const int* child_lr[10]; for (int k=0;k<10;++k) child_lr[k]=I(73+k);
  const int* child_s[9];   for (int k=0;k<9; ++k) child_s[k]=I(83+k);

  static const int nNodes[10] = {512,256,128,64,32,16,8,4,2,1};
  static const int iDim[10]   = {16,32,64,128,128,256,256,512,512,1024};
  static const int outEff[10] = {32,48,96,64,128,128,256,256,512,512};
  auto padK = [](int k){ return (k + 31) & ~31; };

  // --- workspace carve ---
  char* base = (char*)d_ws;
  size_t cur = 0;
  auto alloc    = [&](size_t bytes) -> void* {
    void* p = base + cur;
    cur = (cur + bytes + 255) & ~(size_t)255;
    return p;
  };
  auto allocU16 = [&](size_t n){ return (unsigned short*)alloc(n * 2); };
  auto allocF   = [&](size_t n){ return (float*)alloc(n * 4); };

  // bf16 transposed weights [N, padK(K)]
  unsigned short *wc1=allocU16((size_t)128*32),  *wc2=allocU16((size_t)256*128),
                 *wc3=allocU16((size_t)1024*256);
  unsigned short *wf1=allocU16((size_t)512*1024),*wf2=allocU16((size_t)256*512),
                 *wf3=allocU16((size_t)9*256);
  unsigned short *wl0=allocU16((size_t)64*32),   *wl1=allocU16((size_t)16*64);
  unsigned short* wup[10];
  for (int k=0;k<10;++k) wup[k]=allocU16((size_t)outEff[k]*padK(iDim[k]));

  unsigned short* actA = allocU16((size_t)NP * 256);  // bf16 GEMM input (16MB)
  float* actC   = allocF((size_t)NP * 1024);          // pre-BN GEMM out (128MB)
  float* actD   = allocF((size_t)NP * 256);           // normalized fp32 out (32MB)
  float* mean   = allocF(1024);
  float* invstd = allocF(1024);
  float* gvec   = allocF((size_t)B * 1024);
  float* f3o    = allocF((size_t)B * 9);

  float* tbuf[11];
  int tdim[11], tnodes[11];
  tdim[0]=16; tnodes[0]=1024;
  for (int k=0;k<10;++k) {
    tnodes[k+1]=nNodes[k];
    tdim[k+1]=outEff[k] + (k>=1 ? tdim[k-1] : 0);
  }
  for (int k=0;k<11;++k) tbuf[k]=allocF((size_t)B * tnodes[k] * tdim[k]);

  // --- launch helpers ---
  auto gridFor = [&](long long n){ long long g=(n+WG-1)/WG; return (int)(g>8192?8192:g); };
  auto convert = [&](const float* src, unsigned short* dst, long long n){
    k_f32_to_bf16<<<gridFor(n), WG, 0, stream>>>(src, dst, n);
  };
  auto convertWT = [&](const float* src, unsigned short* dst, int K, int Nc){
    int Kp = padK(K);
    long long tot = (long long)Kp * Nc;
    k_convert_wT<<<gridFor(tot), WG, 0, stream>>>(src, dst, K, Kp, Nc, tot);
  };
  auto gemm = [&](const unsigned short* A, const unsigned short* Wt, const float* bias,
                  float* Cm, int M, int Nc, int Kp){
    dim3 g((Nc + TN - 1)/TN, (M + TM - 1)/TM);
    gemm_bf16_wmma<<<g, dim3(WG), 0, stream>>>(A, Wt, bias, Cm, M, Nc, Kp);
  };
  auto bnstats = [&](const float* X, int M, int C){
    k_bn_stats<<<C, WG, 0, stream>>>(X, mean, invstd, M, C);
  };
  auto bnapply = [&](const float* X, float* Yf, unsigned short* Yb,
                     const float* g_, const float* be_,
                     long long M, int C, int mode, const float* alpha){
    long long tot = M * (long long)C;
    k_bn_apply<<<gridFor(tot), WG, 0, stream>>>(X, Yf, Yb, mean, invstd, g_, be_,
                                                tot, C, mode, alpha);
  };

  // 0) weights -> bf16, transposed to [N, padK(K)]
  convertWT(c1W, wc1, 3, 128);     convertWT(c2W, wc2, 128, 256);
  convertWT(c3W, wc3, 256, 1024);  convertWT(f1W, wf1, 1024, 512);
  convertWT(f2W, wf2, 512, 256);   convertWT(f3W, wf3, 256, 9);
  convertWT(l0W, wl0, 3, 64);      convertWT(l1W, wl1, 64, 16);
  for (int k=0;k<10;++k) convertWT(upW[k], wup[k], iDim[k], outEff[k]);

  // 1) alignment per-point convs (WMMA GEMMs + BN); bf16 output feeds next GEMM
  k_pack_rows_bf16<<<gridFor(NP*32), WG, 0, stream>>>(points, actA, 3, 32, NP*32);
  gemm(actA, wc1, c1b, actC, (int)NP, 128, 32);
  bnstats(actC, (int)NP, 128);
  bnapply(actC, nullptr, actA, c1g, c1be, NP, 128, 1, nullptr);

  gemm(actA, wc2, c2b, actC, (int)NP, 256, 128);
  bnstats(actC, (int)NP, 256);
  bnapply(actC, nullptr, actA, c2g, c2be, NP, 256, 1, nullptr);

  gemm(actA, wc3, c3b, actC, (int)NP, 1024, 256);   // dominant GEMM, 17.2 GFLOP
  bnstats(actC, (int)NP, 1024);
  k_bn_colmax<<<B, WG, 0, stream>>>(actC, gvec, mean, invstd, c3g, c3be, N, 1024);

  // 2) FC head -> 3x3 transform params
  convert(gvec, actA, (long long)B*1024);
  gemm(actA, wf1, f1b, actC, B, 512, 1024);
  bnstats(actC, B, 512);
  bnapply(actC, nullptr, actA, f1g, f1be, B, 512, 1, nullptr);

  gemm(actA, wf2, f2b, actC, B, 256, 512);
  bnstats(actC, B, 256);
  bnapply(actC, nullptr, actA, f2g, f2be, B, 256, 1, nullptr);

  gemm(actA, wf3, f3b, f3o, B, 9, 256);

  // 3) transform + leaf MLP [3 -> 64 (BN,PReLU) -> 16 (BN)]
  k_apply_transform<<<gridFor(NP), WG, 0, stream>>>(points, f3o, actA, N, 32, NP);
  gemm(actA, wl0, l0b, actC, (int)NP, 64, 32);
  bnstats(actC, (int)NP, 64);
  bnapply(actC, nullptr, actA, l0g, l0be, NP, 64, 2, l0a);

  gemm(actA, wl1, l1b, actC, (int)NP, 16, 64);
  bnstats(actC, (int)NP, 16);
  bnapply(actC, actD, nullptr, l1g, l1be, NP, 16, 0, nullptr);

  k_gather_arrange<<<gridFor(NP*16), WG, 0, stream>>>(actD, tbuf[0], arrange, N, 16, NP*16);

  // 4) tree layers: gather -> GEMM -> BN -> max halves (+ sampled concat)
  for (int k=0;k<10;++k) {
    int n = nNodes[k], rows = 2*n, K = iDim[k], Kp = padK(K), OUT = outEff[k];
    long long Mll = (long long)B * rows;
    k_gather_convert<<<gridFor(Mll*Kp), WG, 0, stream>>>(
        tbuf[k], actA, child_lr[k], rows, K, Kp, /*srcNodes=*/rows, Mll*Kp);
    gemm(actA, wup[k], upb[k], actC, (int)Mll, OUT, Kp);
    bnstats(actC, (int)Mll, OUT);
    bnapply(actC, actD, nullptr, upg[k], upbe[k], Mll, OUT, 0, nullptr);

    int sc = (k>=1) ? tdim[k-1] : 0;
    const float* samp = (k>=1) ? tbuf[k-1] : nullptr;
    const int* sidx   = (k>=1) ? child_s[k-1] : nullptr;
    int snodes        = (k>=1) ? tnodes[k-1] : 0;
    long long tot = (long long)B * n * (OUT + sc);
    k_tree_combine<<<gridFor(tot), WG, 0, stream>>>(
        actD, samp, sidx, tbuf[k+1], n, OUT, sc, snodes, tot);
  }

  // 5) root node -> output [B, 1024]
  hipMemcpyAsync(d_out, tbuf[10], (size_t)B * 1024 * sizeof(float),
                 hipMemcpyDeviceToDevice, stream);
}
